// QuantizedLinear_58248346468896
// MI455X (gfx1250) — matrix operand order repaired and checked by reading, not compiled, not run on hardware
//
#include <hip/hip_runtime.h>
#include <hip/hip_bf16.h>
#include <stdint.h>

typedef __attribute__((ext_vector_type(16))) __bf16 v16bf;
typedef __attribute__((ext_vector_type(8)))  __bf16 v8bf;
typedef __attribute__((ext_vector_type(8)))  float  v8f;
typedef __attribute__((ext_vector_type(16))) char   v16c;

#define M_DIM 4096
#define K_DIM 4096
#define N_DIM 8192
#define KT    64
#define NTILES (K_DIM / KT)          // 64 k-tiles

// ---- LDS layout (double buffered) ----
// x tile: 128 rows x 64 bf16 ; w tile: 256 rows x 64 bf16 (dequantized)
#define TSTRIDE 144                  // bytes/row: 64 bf16 = 128B + 16B pad (bank spread)
#define XBUF_SZ (128 * TSTRIDE)      // 18432 B
#define WBUF_SZ (256 * TSTRIDE)      // 36864 B
#define XOFF(b) ((unsigned)((b) * XBUF_SZ))
#define WOFF(b) ((unsigned)(2 * XBUF_SZ + (b) * WBUF_SZ))
#define LDS_TOTAL (2 * XBUF_SZ + 2 * WBUF_SZ)   // 110592 B -> 2 workgroups / 320KB WGP

// CDNA5 async global->LDS copy, tracked with ASYNCcnt. lds_off derives from the
// real shared-mem address so the LDS object escapes into the asm ("memory" clobber
// then makes the asm a visible writer of smem).
__device__ __forceinline__ void async_copy_b128(unsigned lds_off, const void* gaddr) {
  asm volatile("global_load_async_to_lds_b128 %0, %1, off"
               :: "v"(lds_off), "v"((unsigned long long)(uintptr_t)gaddr)
               : "memory");
}
__device__ __forceinline__ void wait_async_all() {
  asm volatile("s_wait_asynccnt 0" ::: "memory");
}

__global__ void __launch_bounds__(256)
qlinear_bf16_wmma(const __hip_bfloat16* __restrict__ x,
                  const int8_t*        __restrict__ wq,
                  const float*         __restrict__ wscale,
                  const __hip_bfloat16* __restrict__ bias,
                  __hip_bfloat16*       __restrict__ out) {
  extern __shared__ __align__(16) unsigned char smem[];   // dynamic LDS: 110592 B

  const int tid  = threadIdx.x;
  const int lane = tid & 31;
  const int wave = tid >> 5;
  const int wm   = wave & 1;           // 0..1 : 64-row M slice
  const int wn   = wave >> 1;          // 0..3 : 64-col N slice
  const int n0   = blockIdx.x * 256;   // spans TWO 128-wide scale blocks
  const int m0   = blockIdx.y * 128;

  // Low 32 bits of the flat shared address == LDS byte offset (aperture rule).
  const unsigned lds_base = (unsigned)(uintptr_t)(void*)smem;

  const char* xg = (const char*)x;
  const char* wg = (const char*)wq;

  // ---- async x tile: 128x64 bf16 = 1024 x 16B chunks (4 per thread) ----
  auto issue_x = [&](int t, int buf) {
    const long xrow0 = (long)m0 * (K_DIM * 2);
    const long kbx   = (long)t * (KT * 2);
#pragma unroll
    for (int i = 0; i < 4; ++i) {
      int chunk = tid + i * 256;
      int r = chunk >> 3, c = chunk & 7;
      async_copy_b128(lds_base + XOFF(buf) + r * TSTRIDE + c * 16,
                      xg + xrow0 + (long)r * (K_DIM * 2) + kbx + c * 16);
    }
  };

  // ---- W tile: 256x64 int8 = 1024 x 16B chunks (4 per thread), regs in flight ----
  auto load_w = [&](int t, v16c wr[4]) {
    const long wrow0 = (long)n0 * K_DIM;
    const long kbw   = (long)t * KT;
#pragma unroll
    for (int i = 0; i < 4; ++i) {
      int chunk = tid + i * 256;
      int r = chunk >> 2, c = chunk & 3;
      wr[i] = *(const v16c*)(wg + wrow0 + (long)r * K_DIM + kbw + c * 16);
    }
  };

  // ---- dequant once per tile: int8 regs -> bf16 LDS (scale folded in) ----
  // chunk groups i=0,1 -> rows 0..127 (scale s.x) ; i=2,3 -> rows 128..255 (s.y)
  auto stage_w = [&](const v16c wr[4], int buf, float s_lo, float s_hi) {
#pragma unroll
    for (int i = 0; i < 4; ++i) {
      const float s = (i < 2) ? s_lo : s_hi;
      int chunk = tid + i * 256;
      int r = chunk >> 2, c = chunk & 3;     // 16 elements -> 32 bytes of bf16
      v16bf b;
#pragma unroll
      for (int j = 0; j < 16; ++j)
        b[j] = (__bf16)((float)(signed char)wr[i][j] * s);
      unsigned char* dst = smem + WOFF(buf) + r * TSTRIDE + c * 32;
      *(v8bf*)(dst)      = __builtin_shufflevector(b, b, 0,1,2,3,4,5,6,7);
      *(v8bf*)(dst + 16) = __builtin_shufflevector(b, b, 8,9,10,11,12,13,14,15);
    }
  };

  v8f acc[4][4] = {};                  // 4 m-frags x 4 n-frags of 16x16 f32
  v16c wr[4];

  const int sb = blockIdx.x * 2 * (K_DIM / 128);        // scale row base (two n-blocks)

  // ---- prologue: tile 0 ----
  load_w(0, wr);
  issue_x(0, 0);
  stage_w(wr, 0, wscale[sb], wscale[sb + (K_DIM / 128)]);
  wait_async_all();
  __syncthreads();

  for (int t = 0; t < NTILES; ++t) {
    const int buf = t & 1;
    if (t + 1 < NTILES) {
      load_w(t + 1, wr);               // LOADcnt in flight across the WMMAs below
      issue_x(t + 1, buf ^ 1);
    }

    const unsigned char* xt = smem + XOFF(buf);
    const unsigned char* wt = smem + WOFF(buf);

#pragma unroll
    for (int ks = 0; ks < 2; ++ks) {   // two 16x16x32 k-steps per 64-K tile
      // ---- phase 1: load all 8 fragments for this k-step ----
      v16bf a[4];
#pragma unroll
      for (int mf = 0; mf < 4; ++mf) { // A: lane m rows; half-wave K split 0..7 / 8..15
        const int row = wm * 64 + mf * 16 + (lane & 15);
        const int klo = ks * 32 + ((lane >> 4) << 3);     // 0 or 8
        const v8bf lo = *(const v8bf*)(xt + row * TSTRIDE + klo * 2);
        const v8bf hi = *(const v8bf*)(xt + row * TSTRIDE + klo * 2 + 32);
        a[mf] = __builtin_shufflevector(lo, hi, 0,1,2,3,4,5,6,7,8,9,10,11,12,13,14,15);
      }
      v16bf b[4];
#pragma unroll
      for (int nf = 0; nf < 4; ++nf) { // B: lane n holds w[n, k..k+7] and w[n, k+16..k+23] (pre-dequantized)
        const int col  = wn * 64 + nf * 16 + (lane & 15);
        const int koff = ks * 32 + ((lane >> 4) << 3);    // 0 or 8
        const v8bf lo = *(const v8bf*)(wt + col * TSTRIDE + koff * 2);
        const v8bf hi = *(const v8bf*)(wt + col * TSTRIDE + koff * 2 + 32);
        b[nf] = __builtin_shufflevector(lo, hi, 0,1,2,3,4,5,6,7,8,9,10,11,12,13,14,15);
      }
      // ---- phase 2: 16 back-to-back WMMAs (1.0 ds_load per wmma) ----
#pragma unroll
      for (int nf = 0; nf < 4; ++nf)
#pragma unroll
        for (int mf = 0; mf < 4; ++mf)
          acc[mf][nf] = __builtin_amdgcn_wmma_f32_16x16x32_bf16(
              false, a[mf], false, b[nf], (short)0, acc[mf][nf], false, false);
    }

    if (t + 1 < NTILES) {
      // dequant tile t+1 into the other buffer; its previous readers finished at
      // the last barrier, current compute reads `buf` only.
      const int kb = (t + 1) >> 1;
      stage_w(wr, buf ^ 1, wscale[sb + kb], wscale[sb + (K_DIM / 128) + kb]);
    }

    wait_async_all();                  // x tile t+1 landed
    __syncthreads();                   // + everyone done reading tile t / staging done
  }

  // Epilogue: C/D layout -> VGPR i is row m+i (lanes 16-31: m+8+i), lanes are columns
#pragma unroll
  for (int nf = 0; nf < 4; ++nf) {
    const int ncol = n0 + wn * 64 + nf * 16 + (lane & 15);
    const float bf = __bfloat162float(bias[ncol]);
#pragma unroll
    for (int mf = 0; mf < 4; ++mf) {
      const int mrow = m0 + wm * 64 + mf * 16 + ((lane >> 4) << 3);
#pragma unroll
      for (int i = 0; i < 8; ++i)
        out[(long)(mrow + i) * N_DIM + ncol] = __float2bfloat16(acc[mf][nf][i] + bf);
    }
  }
}

extern "C" void kernel_launch(void* const* d_in, const int* /*in_sizes*/, int /*n_in*/,
                              void* d_out, int /*out_size*/, void* /*d_ws*/, size_t /*ws_size*/,
                              hipStream_t stream) {
  const __hip_bfloat16* x   = (const __hip_bfloat16*)d_in[0];
  const int8_t*         wq  = (const int8_t*)d_in[1];
  const float*          ws  = (const float*)d_in[2];
  const __hip_bfloat16* bia = (const __hip_bfloat16*)d_in[3];
  __hip_bfloat16*       out = (__hip_bfloat16*)d_out;

  dim3 grid(N_DIM / 256, M_DIM / 128);   // 32 x 32 workgroups
  qlinear_bf16_wmma<<<grid, dim3(256), LDS_TOTAL, stream>>>(x, wq, ws, bia, out);
}
